// SelfAttention_3126736191503
// MI455X (gfx1250) — matrix operand order repaired and checked by reading, not compiled, not run on hardware
//
#include <hip/hip_runtime.h>

// ---------------------------------------------------------------------------
// Problem constants (from the reference): B=8, L=4096, D=1024, H=16, DK=64
// ---------------------------------------------------------------------------
#define NPOS   32768          // B * L
#define DIM    1024           // D

typedef __bf16 bf16_t;
typedef __attribute__((ext_vector_type(16))) __bf16 v16bf;
typedef __attribute__((ext_vector_type(8)))  float  v8f;

// f32 -> bf16 (native RNE convert, v_cvt_*bf16*)
static __device__ __forceinline__ bf16_t f2bf(float f) { return (bf16_t)f; }

// ---------------------------------------------------------------------------
// CDNA5 async global->LDS copy (16B per lane), with synchronous fallback.
// Builtin param0 (from hipcc diagnostic): non-const AS(1) pointer to
// int __attribute__((vector_size(16))); LDS side AS(3).
// ---------------------------------------------------------------------------
#if __has_builtin(__builtin_amdgcn_global_load_async_to_lds_b128)
#define HAVE_ASYNC_LDS 1
#else
#define HAVE_ASYNC_LDS 0
#endif

typedef int v4i_vs __attribute__((vector_size(16)));
#define AS1P __attribute__((address_space(1)))
#define AS3P __attribute__((address_space(3)))

static __device__ __forceinline__ void async_copy16(const bf16_t* g, bf16_t* l) {
#if HAVE_ASYNC_LDS
    __builtin_amdgcn_global_load_async_to_lds_b128(
        (AS1P v4i_vs*)(uintptr_t)g,
        (AS3P v4i_vs*)(uintptr_t)l, 0, 0);
#else
    *(uint4*)l = *(const uint4*)g;
#endif
}

static __device__ __forceinline__ void async_wait0() {
#if HAVE_ASYNC_LDS
#if __has_builtin(__builtin_amdgcn_s_wait_asynccnt)
    __builtin_amdgcn_s_wait_asynccnt(0);
#else
    asm volatile("s_wait_asynccnt 0x0" ::: "memory");
#endif
#endif
}

// ---------------------------------------------------------------------------
// Kernel 1: f32 -> bf16 conversion (with optional scale), 8 elems / thread.
// The 1/sqrt(dk) softmax scale is folded into wq here (exact: power of two).
// ---------------------------------------------------------------------------
__global__ __launch_bounds__(256) void k_cvt_bf16(const float* __restrict__ src,
                                                  bf16_t* __restrict__ dst,
                                                  long n8, float scale) {
    long i = (long)blockIdx.x * blockDim.x + threadIdx.x;
    if (i >= n8) return;
    const float4* s = (const float4*)src;
    float4 a = s[2 * i];
    float4 b = s[2 * i + 1];
    union { bf16_t h[8]; uint4 q; } u;
    u.h[0] = f2bf(a.x * scale); u.h[1] = f2bf(a.y * scale);
    u.h[2] = f2bf(a.z * scale); u.h[3] = f2bf(a.w * scale);
    u.h[4] = f2bf(b.x * scale); u.h[5] = f2bf(b.y * scale);
    u.h[6] = f2bf(b.z * scale); u.h[7] = f2bf(b.w * scale);
    ((uint4*)dst)[i] = u.q;
}

// ---------------------------------------------------------------------------
// Kernel 2/4: NT GEMM  O[n, m] = sum_k A[n,k] * W[m,k]   (A,W bf16, acc f32)
//   block tile 128x128, 8 waves, wave tile 64x32 (4x2 WMMA accs), K-step 32,
//   double-buffered LDS fed by global_load_async_to_lds_b128.
// ---------------------------------------------------------------------------
template <bool OUT_BF16>
__global__ __launch_bounds__(256) void k_gemm_nt(const bf16_t* __restrict__ A,
                                                 const bf16_t* __restrict__ Wbase,
                                                 void* __restrict__ Obase,
                                                 int M, int K, int Nd) {
    const int mat = blockIdx.z;
    const bf16_t* W = Wbase + (size_t)mat * Nd * K;

    const int bm = blockIdx.x * 128;
    const int bn = blockIdx.y * 128;

    __shared__ bf16_t sA[2][128][32];
    __shared__ bf16_t sB[2][128][32];

    const int tid  = threadIdx.x;
    const int lane = tid & 31;
    const int wid  = tid >> 5;
    const int wm   = (wid >> 2) * 64;
    const int wn   = (wid & 3) * 32;
    const bool hi  = lane >= 16;
    const int l16  = lane & 15;

    v8f acc[4][2];
    v8f zero = {};
#pragma unroll
    for (int i = 0; i < 4; ++i)
#pragma unroll
        for (int j = 0; j < 2; ++j) acc[i][j] = zero;

    const int ldrow = tid >> 1;
    const int ldkh  = (tid & 1) * 16;

    auto load_tile = [&](int kk, int buf) {
        const bf16_t* ga = A + (size_t)(bm + ldrow) * K + (kk + ldkh);
        const bf16_t* gb = W + (size_t)(bn + ldrow) * K + (kk + ldkh);
        async_copy16(ga,     &sA[buf][ldrow][ldkh]);
        async_copy16(ga + 8, &sA[buf][ldrow][ldkh + 8]);
        async_copy16(gb,     &sB[buf][ldrow][ldkh]);
        async_copy16(gb + 8, &sB[buf][ldrow][ldkh + 8]);
    };

    int cur = 0;
    load_tile(0, 0);

    for (int kk = 0; kk < K; kk += 32) {
        async_wait0();          // my slice of buf[cur] landed
        __syncthreads();        // everyone's slice landed; prev reads done
        if (kk + 32 < K) load_tile(kk + 32, cur ^ 1);   // DMA next tile

        // fragments (ISA 7.12.2 16-bit layouts), all b128 LDS loads
        v16bf aF[4], bF[2];
#pragma unroll
        for (int ti = 0; ti < 4; ++ti) {
            int r = wm + ti * 16 + l16;
            uint4* d = (uint4*)&aF[ti];
            d[0] = *(const uint4*)&sA[cur][r][hi ? 8 : 0];
            d[1] = *(const uint4*)&sA[cur][r][16 + (hi ? 8 : 0)];
        }
#pragma unroll
        for (int tj = 0; tj < 2; ++tj) {
            int c = wn + tj * 16 + l16;
            uint4* d = (uint4*)&bF[tj];
            d[0] = *(const uint4*)&sB[cur][c][(hi ? 8 : 0)];
            d[1] = *(const uint4*)&sB[cur][c][16 + (hi ? 8 : 0)];
        }

#pragma unroll
        for (int ti = 0; ti < 4; ++ti)
#pragma unroll
            for (int tj = 0; tj < 2; ++tj)
                acc[ti][tj] = __builtin_amdgcn_wmma_f32_16x16x32_bf16(
                    false, aF[ti], false, bF[tj], (short)0, acc[ti][tj], false, false);

        cur ^= 1;
    }

    // store (C layout: VGPR r -> M=r (lanes 0-15) / M=8+r (lanes 16-31))
#pragma unroll
    for (int ti = 0; ti < 4; ++ti)
#pragma unroll
        for (int tj = 0; tj < 2; ++tj) {
            int gm = bm + wm + ti * 16 + (hi ? 8 : 0);
            int gn = bn + wn + tj * 16 + l16;
            if (OUT_BF16) {
                bf16_t* O = (bf16_t*)Obase + (size_t)mat * M * Nd;
#pragma unroll
                for (int r = 0; r < 8; ++r)
                    O[(size_t)(gm + r) * Nd + gn] = f2bf(acc[ti][tj][r]);
            } else {
                float* O = (float*)Obase;
#pragma unroll
                for (int r = 0; r < 8; ++r)
                    O[(size_t)(gm + r) * Nd + gn] = acc[ti][tj][r];
            }
        }
}

// ---------------------------------------------------------------------------
// Kernel 3: per-position attention.  One wave32 per token position.
//   Q_l,K_l,V_l are [DK=64, H=16] (row stored as [h*64+dk]); Q is pre-scaled
//   by 1/sqrt(dk) (folded into wq), so scores need no scaling here.
//   scores = Q K^T (64x64, K=H=16 zero-padded to 32 *in LDS*),
//   softmax without max-subtraction (scores are bounded),
//   row sums via WMMA against all-ones B (no cross-lane shuffles),
//   out = (E@V) * rcp(E@1) -> concat row [h*64+i].
//
//   Per-wave LDS arena (10KB), staging region overlaid by the E matrix:
//     [0   ..2047]  sQt[64][32]  (pad h=16..31 zeroed)   -+ later: sAttn[64][64]
//     [2048..4095]  sKt[64][32]  (pad h=16..31 zeroed)   -+        (8KB)
//     [4096..5119]  sV[16][64]
// ---------------------------------------------------------------------------
__global__ __launch_bounds__(128) void k_attn(const bf16_t* __restrict__ Q,
                                              const bf16_t* __restrict__ Kv,
                                              const bf16_t* __restrict__ V,
                                              bf16_t* __restrict__ Cc) {
    __shared__ __align__(16) bf16_t arena[4][5120];   // 40KB / block

    const int wid  = threadIdx.x >> 5;
    const int lane = threadIdx.x & 31;
    const bool hi  = lane >= 16;
    const int l16  = lane & 15;
    const long pos = (long)blockIdx.x * 4 + wid;

    bf16_t* sQt   = &arena[wid][0];       // [64][32], cols 16..31 = 0
    bf16_t* sKt   = &arena[wid][2048];    // [64][32], cols 16..31 = 0
    bf16_t* sAttn = &arena[wid][0];       // [64][64], overlays sQt+sKt
    bf16_t* sV    = &arena[wid][4096];    // [16][64]

    // --- V: async DMA straight into LDS (natural [h][dk] layout) ---
    {
        const bf16_t* gv = V + pos * DIM;
#pragma unroll
        for (int t = 0; t < 4; ++t) {
            int e8 = (lane + 32 * t) * 8;
            async_copy16(gv + e8, sV + e8);
        }
    }
    // --- Q,K: coalesced b128 global loads, register-side transpose into LDS,
    //     plus zero-fill of the h=16..31 pad columns ---
    {
        const uint4 z4 = {};
#pragma unroll
        for (int t = 0; t < 2; ++t) {     // zero pads: 64 rows x 32B each
            int r = lane + 32 * t;
            *(uint4*)&sQt[r * 32 + 16] = z4;
            *(uint4*)&sQt[r * 32 + 24] = z4;
            *(uint4*)&sKt[r * 32 + 16] = z4;
            *(uint4*)&sKt[r * 32 + 24] = z4;
        }
        const uint4* gq = (const uint4*)(Q  + pos * DIM);
        const uint4* gk = (const uint4*)(Kv + pos * DIM);
#pragma unroll
        for (int t = 0; t < 4; ++t) {
            int e  = lane + 32 * t;       // uint4 index; covers elems e*8..e*8+7
            int h  = (e * 8) >> 6;        // head index
            int d0 = (e * 8) & 63;        // dk start
            union { uint4 q; bf16_t h8[8]; } uq, uk;
            uq.q = gq[e];
            uk.q = gk[e];
#pragma unroll
            for (int x = 0; x < 8; ++x) {
                sQt[(d0 + x) * 32 + h] = uq.h8[x];
                sKt[(d0 + x) * 32 + h] = uk.h8[x];
            }
        }
    }
    async_wait0();
    __syncthreads();

    // --- fragments: pure b128 loads, zero-padding comes from LDS ---
    v16bf qF[4];                 // A tile [dk x h(pad32)]
#pragma unroll
    for (int ti = 0; ti < 4; ++ti) {
        int i = ti * 16 + l16;
        uint4* d = (uint4*)&qF[ti];
        d[0] = *(const uint4*)&sQt[i * 32 + (hi ? 8 : 0)];
        d[1] = *(const uint4*)&sQt[i * 32 + 16 + (hi ? 8 : 0)];   // zeros
    }
    v16bf kF[4];                 // B tile [h(pad32) x j]
#pragma unroll
    for (int tj = 0; tj < 4; ++tj) {
        int j = tj * 16 + l16;
        uint4* d = (uint4*)&kF[tj];
        d[0] = *(const uint4*)&sKt[j * 32 + (hi ? 8 : 0)];
        d[1] = *(const uint4*)&sKt[j * 32 + 16 + (hi ? 8 : 0)];   // zeros
    }
    __syncthreads();   // fragments in VGPRs; staging region may be overwritten

    // --- scores row-by-row (only 4 accumulators live), E -> sAttn overlay ---
    v8f zero = {};
#pragma unroll
    for (int ti = 0; ti < 4; ++ti) {
        v8f scRow[4];
#pragma unroll
        for (int tj = 0; tj < 4; ++tj)
            scRow[tj] = __builtin_amdgcn_wmma_f32_16x16x32_bf16(
                false, qF[ti], false, kF[tj], (short)0, zero, false, false);
#pragma unroll
        for (int r = 0; r < 8; ++r) {
            int i = ti * 16 + (hi ? 8 + r : r);
#pragma unroll
            for (int tj = 0; tj < 4; ++tj)
                sAttn[i * 64 + tj * 16 + l16] = f2bf(__expf(scRow[tj][r]));
        }
    }
    __syncthreads();

    // --- out = E @ V, rowsum = E @ ones (same C layout => per-lane normalize) ---
    v16bf vF[2];
#pragma unroll
    for (int jc = 0; jc < 2; ++jc) {
        const bf16_t* p = &sV[l16 * 64 + jc * 32 + (hi ? 8 : 0)];
        uint4* d = (uint4*)&vF[jc];
        d[0] = ((const uint4*)p)[0];
        d[1] = ((const uint4*)p)[2];
    }
    v16bf oneF;
    {
        bf16_t one = __builtin_bit_cast(bf16_t, (unsigned short)0x3F80);
#pragma unroll
        for (int x = 0; x < 16; ++x) oneF[x] = one;
    }
    v8f oacc[4], sacc[4];
#pragma unroll
    for (int ti = 0; ti < 4; ++ti) { oacc[ti] = zero; sacc[ti] = zero; }
#pragma unroll
    for (int ti = 0; ti < 4; ++ti)
#pragma unroll
        for (int jc = 0; jc < 2; ++jc) {
            v16bf aF;
            const bf16_t* p = &sAttn[(ti * 16 + l16) * 64 + jc * 32 + (hi ? 8 : 0)];
            uint4* d = (uint4*)&aF;
            d[0] = *(const uint4*)p;
            d[1] = *(const uint4*)(p + 16);
            oacc[ti] = __builtin_amdgcn_wmma_f32_16x16x32_bf16(
                false, aF, false, vF[jc], (short)0, oacc[ti], false, false);
            sacc[ti] = __builtin_amdgcn_wmma_f32_16x16x32_bf16(
                false, aF, false, oneF, (short)0, sacc[ti], false, false);
        }
    __syncthreads();

    // --- normalize, scatter to row buffer (reuse overlay), coalesced store ---
    bf16_t* orow = sAttn;                // 1024 bf16 = concat row [h*64 + i]
#pragma unroll
    for (int ti = 0; ti < 4; ++ti)
#pragma unroll
        for (int r = 0; r < 8; ++r) {
            int i = ti * 16 + (hi ? 8 + r : r);
            float inv = __builtin_amdgcn_rcpf(sacc[ti][r]);
            orow[l16 * 64 + i] = f2bf(oacc[ti][r] * inv);
        }
    __syncthreads();
    {
        uint4* gc = (uint4*)(Cc + pos * DIM);
        const uint4* lr = (const uint4*)orow;
#pragma unroll
        for (int t = 0; t < 4; ++t) gc[lane + 32 * t] = lr[lane + 32 * t];
    }
}

// ---------------------------------------------------------------------------
// Host-side launch (graph-capture safe: only kernel launches on `stream`)
// Workspace layout (bf16 scratch, ~344 MB):
//   xb[N*D] | wqkv[3*D*D] | wo[D*D] | qkv[3*N*D] | concat[N*D]
// ---------------------------------------------------------------------------
extern "C" void kernel_launch(void* const* d_in, const int* in_sizes, int n_in,
                              void* d_out, int out_size, void* d_ws, size_t ws_size,
                              hipStream_t stream) {
    (void)in_sizes; (void)n_in; (void)out_size; (void)ws_size;
    const float* x  = (const float*)d_in[0];
    const float* wq = (const float*)d_in[1];
    const float* wk = (const float*)d_in[2];
    const float* wv = (const float*)d_in[3];
    const float* wo = (const float*)d_in[4];

    const long ND = (long)NPOS * DIM;
    const long DD = (long)DIM * DIM;

    char* ws = (char*)d_ws;
    size_t off = 0;
    bf16_t* xb     = (bf16_t*)(ws + off); off += (size_t)ND * 2;
    bf16_t* wqkvb  = (bf16_t*)(ws + off); off += (size_t)3 * DD * 2;
    bf16_t* wob    = (bf16_t*)(ws + off); off += (size_t)DD * 2;
    bf16_t* qkv    = (bf16_t*)(ws + off); off += (size_t)3 * ND * 2;
    bf16_t* concat = (bf16_t*)(ws + off); off += (size_t)ND * 2;

    // 1) convert inputs to bf16; fold softmax 1/sqrt(dk)=0.125 into wq (exact)
    {
        long n8 = ND / 8;
        k_cvt_bf16<<<(unsigned)((n8 + 255) / 256), 256, 0, stream>>>(x, xb, n8, 1.0f);
        long w8 = DD / 8;
        unsigned wb = (unsigned)((w8 + 255) / 256);
        k_cvt_bf16<<<wb, 256, 0, stream>>>(wq, wqkvb,          w8, 0.125f);
        k_cvt_bf16<<<wb, 256, 0, stream>>>(wk, wqkvb + DD,     w8, 1.0f);
        k_cvt_bf16<<<wb, 256, 0, stream>>>(wv, wqkvb + 2 * DD, w8, 1.0f);
        k_cvt_bf16<<<wb, 256, 0, stream>>>(wo, wob,            w8, 1.0f);
    }

    // 2) Q/K/V projections: gridDim.z selects the weight matrix
    {
        dim3 grid(NPOS / 128, DIM / 128, 3);
        k_gemm_nt<true><<<grid, 256, 0, stream>>>(xb, wqkvb, (void*)qkv,
                                                  NPOS, DIM, DIM);
    }

    // 3) per-position attention -> concat (bf16)
    k_attn<<<NPOS / 4, 128, 0, stream>>>(qkv, qkv + ND, qkv + 2 * ND, concat);

    // 4) output projection -> d_out (f32)
    {
        dim3 grid(NPOS / 128, DIM / 128, 1);
        k_gemm_nt<false><<<grid, 256, 0, stream>>>(concat, wob, d_out,
                                                   NPOS, DIM, DIM);
    }
}
